// Elmo_37503654429058
// MI455X (gfx1250) — matrix operand order repaired and checked by reading, not compiled, not run on hardware
//
#include <hip/hip_runtime.h>
#include <hip/hip_bf16.h>

// ---------------------------------------------------------------------------
// ELMo biLSTM for MI455X (gfx1250, wave32, WMMA).
//   B=64, T=512, H=1024, gates G=4H=4096, ninp=512.
//   Fuse x@Wih^T + h@Whh^T into one per-timestep GEMM over K = I + H against
//   a packed bf16 [Wih | Whh] (L2-resident, 12-17 MB), accumulate fp32 via
//   v_wmma_f32_16x16x32_bf16, recombine i/f/g/o through LDS, fp32 cell.
//   Fragment loads are forced to address_space(1) so they lower to
//   global_load_b128 (LOADcnt only), not flat_load (LOADcnt+DScnt).
// ---------------------------------------------------------------------------

typedef __bf16 bf16;
typedef __attribute__((ext_vector_type(16))) __bf16 v16bf;
typedef __attribute__((ext_vector_type(8)))  __bf16 v8bf;
typedef __attribute__((ext_vector_type(8)))  float  v8f;

union V16 { v16bf v; v8bf h[2]; };

// Global-address-space vector load: guarantees global_load_b128.
typedef const __attribute__((address_space(1))) v8bf* gv8bf_p;
__device__ __forceinline__ v8bf gld8(const bf16* p) {
    return *(gv8bf_p)(unsigned long long)p;
}

#define TSEQ 512
#define BATCH 64
#define HID  1024
#define GATES 4096
#define NINP 512

__device__ __forceinline__ float sigm_f(float x) {
    return 1.0f / (1.0f + __expf(-x));
}
__device__ __forceinline__ float tanh_f(float x) {
    float e = __expf(-2.0f * fabsf(x));
    float r = (1.0f - e) / (1.0f + e);
    return copysignf(r, x);
}

// ---------------------------------------------------------------------------
// Embedding gather: emb_f[t,b,:] = enc[input[b,t]],
//                   emb_r[t,b,:] = enc[input[b, (t<L)? L-1-t : t]]  (bf16)
// ---------------------------------------------------------------------------
__global__ void embed_gather(const int* __restrict__ tok,
                             const int* __restrict__ lens,
                             const float* __restrict__ enc,
                             bf16* __restrict__ ef, bf16* __restrict__ er) {
    int tb = blockIdx.x;             // T*B blocks
    int t = tb >> 6, b = tb & 63;
    int L = lens[b];
    int tf = tok[b * TSEQ + t];
    int tri = (t < L) ? (L - 1 - t) : t;
    int trk = tok[b * TSEQ + tri];
    const float* sf = enc + (size_t)tf  * NINP;
    const float* sr = enc + (size_t)trk * NINP;
    size_t dst = ((size_t)t * BATCH + b) * NINP;
    for (int i = threadIdx.x; i < NINP; i += blockDim.x) {
        ef[dst + i] = (bf16)sf[i];
        er[dst + i] = (bf16)sr[i];
    }
}

// ---------------------------------------------------------------------------
// Pack Wcat[n, k] = (k < K_in) ? Wih[n,k] : Whh[n,k-K_in]  (bf16), and
// bias[n] = bih[n] + bhh[n] (fp32).
// ---------------------------------------------------------------------------
__global__ void pack_weights(const float* __restrict__ Wih,
                             const float* __restrict__ Whh,
                             const float* __restrict__ bih,
                             const float* __restrict__ bhh,
                             bf16* __restrict__ Wcat, float* __restrict__ bias,
                             int K_in) {
    int Kt = K_in + HID;
    size_t total = (size_t)GATES * Kt;
    size_t stride = (size_t)gridDim.x * blockDim.x;
    for (size_t idx = blockIdx.x * (size_t)blockDim.x + threadIdx.x;
         idx < total; idx += stride) {
        size_t n = idx / Kt;
        int k = (int)(idx % Kt);
        float v = (k < K_in) ? Wih[n * K_in + k] : Whh[n * HID + (k - K_in)];
        Wcat[idx] = (bf16)v;
    }
    size_t gi = blockIdx.x * (size_t)blockDim.x + threadIdx.x;
    if (gi < GATES) bias[gi] = bih[gi] + bhh[gi];
}

// ---------------------------------------------------------------------------
// Zero h (bf16 ping-pong, both dirs) and c (fp32, both dirs).
// ---------------------------------------------------------------------------
__global__ void init_state(bf16* h0f, bf16* h1f, bf16* h0r, bf16* h1r,
                           float* cf, float* cr) {
    int i = blockIdx.x * blockDim.x + threadIdx.x;
    if (i < BATCH * HID) {
        bf16 z = (bf16)0.0f;
        h0f[i] = z; h1f[i] = z; h0r[i] = z; h1r[i] = z;
        cf[i] = 0.0f; cr[i] = 0.0f;
    }
}

// ---------------------------------------------------------------------------
// One LSTM timestep, both directions (blockIdx.y = dir).
// Grid (64, 2), block = 128 threads = 4 waves.
//   blockIdx.x -> 16 hidden units u0..u0+15
//   wave w     -> gate w, GEMM N-strip = cols [w*1024+u0, +16)
//   each wave computes 4 M-tiles (all 64 batch rows) x 16 cols, K = K_in+1024.
// ---------------------------------------------------------------------------

// GEMM phase over one contiguous A region: bumped-pointer K loop.
__device__ __forceinline__ void gemm_phase(const bf16* __restrict__ abase,
                                           int astride, int klen,
                                           const bf16*& wp,
                                           int lrow, int lhalf,
                                           v8f acc[4]) {
    const bf16* ap[4];
#pragma unroll
    for (int mt = 0; mt < 4; ++mt)
        ap[mt] = abase + (size_t)(mt * 16 + lrow) * astride + lhalf * 16;

    for (int k0 = 0; k0 < klen; k0 += 32) {
        // weight-stream prefetch ahead (global_prefetch_b8)
        __builtin_prefetch(wp + 512, 0, 1);

        // B fragment: col = lane's W row, K = lhalf*16 + [0..15] (contiguous)
        V16 bfrag;
        bfrag.h[0] = gld8(wp);
        bfrag.h[1] = gld8(wp + 8);
        wp += 32;

#pragma unroll
        for (int mt = 0; mt < 4; ++mt) {
            // A fragment: elems 0..15 -> K=lhalf*16+i (contiguous, matches B)
            V16 afrag;
            afrag.h[0] = gld8(ap[mt]);
            afrag.h[1] = gld8(ap[mt] + 8);
            ap[mt] += 32;
            acc[mt] = __builtin_amdgcn_wmma_f32_16x16x32_bf16(
                false, afrag.v, false, bfrag.v, (short)0, acc[mt], false, false);
        }
    }
}

__global__ __launch_bounds__(128)
void lstm_step(const bf16* __restrict__ x_f, const bf16* __restrict__ x_r,
               const bf16* __restrict__ W_f, const bf16* __restrict__ W_r,
               const float* __restrict__ bias_f, const float* __restrict__ bias_r,
               const bf16* __restrict__ hp_f, const bf16* __restrict__ hp_r,
               bf16* __restrict__ hn_f, bf16* __restrict__ hn_r,
               float* __restrict__ c_f, float* __restrict__ c_r,
               const int* __restrict__ lens, int t, int K_in,
               bf16* __restrict__ xo_f, bf16* __restrict__ xo_r,
               float* __restrict__ dout) {
    const int dir = blockIdx.y;
    const bf16*  xbase = dir ? x_r   : x_f;
    const bf16*  W     = dir ? W_r   : W_f;
    const float* bias  = dir ? bias_r: bias_f;
    const bf16*  hp    = dir ? hp_r  : hp_f;
    bf16*        hn    = dir ? hn_r  : hn_f;
    float*       cc    = dir ? c_r   : c_f;
    bf16*        xo    = dir ? xo_r  : xo_f;

    const int Kt    = K_in + HID;
    const int wave  = threadIdx.x >> 5;   // gate index 0..3 (i,f,g,o)
    const int lane  = threadIdx.x & 31;
    const int lrow  = lane & 15;
    const int lhalf = lane >> 4;
    const int u0    = blockIdx.x * 16;
    const int n0    = wave * HID + u0;    // GEMM column base for this wave

    const bf16* xt = xbase + (size_t)t * BATCH * K_in;
    // one W row per lane (K-contiguous), pre-offset by lane's K subgroup
    const bf16* wp = W + (size_t)(n0 + lrow) * Kt + lhalf * 16;

    v8f acc[4] = {};                      // 4 M-tiles x (8 fp32 / lane)

    gemm_phase(xt, K_in, K_in, wp, lrow, lhalf, acc);  // x part of fused K
    gemm_phase(hp, HID,  HID,  wp, lrow, lhalf, acc);  // h part of fused K

    // Recombine i/f/g/o tiles in LDS: gbuf[gate][batch_row][unit]
    __shared__ float gbuf[4][BATCH][16];
    const float bias_c = bias[n0 + lrow];
#pragma unroll
    for (int mt = 0; mt < 4; ++mt)
#pragma unroll
        for (int r = 0; r < 8; ++r)
            gbuf[wave][mt * 16 + lhalf * 8 + r][lrow] = acc[mt][r] + bias_c;
    __syncthreads();

    // Elementwise LSTM cell: 64x16 units, 8 per thread
#pragma unroll
    for (int j = 0; j < 8; ++j) {
        int e = threadIdx.x + j * 128;
        int b = e >> 4, u = e & 15;
        int uu = u0 + u;
        float iv = sigm_f(gbuf[0][b][u]);
        float fv = sigm_f(gbuf[1][b][u]);
        float gv = tanh_f(gbuf[2][b][u]);
        float ov = sigm_f(gbuf[3][b][u]);
        float cold = cc[b * HID + uu];
        float cn = fv * cold + iv * gv;
        float hnew = ov * tanh_f(cn);
        bool  m = t < lens[b];
        float hcar = m ? hnew : (float)hp[b * HID + uu];
        cc[b * HID + uu] = m ? cn : cold;
        hn[b * HID + uu] = (bf16)hcar;
        float ho = m ? hnew : 0.0f;
        if (dout) {
            dout[((size_t)t * BATCH + b) * 2048 + (size_t)dir * HID + uu] = ho;
        } else {
            xo[((size_t)t * BATCH + b) * HID + uu] = (bf16)ho;
        }
    }
}

// ---------------------------------------------------------------------------
// Host launcher
// ---------------------------------------------------------------------------
extern "C" void kernel_launch(void* const* d_in, const int* in_sizes, int n_in,
                              void* d_out, int out_size, void* d_ws, size_t ws_size,
                              hipStream_t stream) {
    (void)in_sizes; (void)n_in; (void)out_size; (void)ws_size;

    const int*   tok  = (const int*)d_in[0];
    const int*   lens = (const int*)d_in[1];
    const float* enc  = (const float*)d_in[2];
    const float* Wih_f0 = (const float*)d_in[3],  *Whh_f0 = (const float*)d_in[4];
    const float* bih_f0 = (const float*)d_in[5],  *bhh_f0 = (const float*)d_in[6];
    const float* Wih_f1 = (const float*)d_in[7],  *Whh_f1 = (const float*)d_in[8];
    const float* bih_f1 = (const float*)d_in[9],  *bhh_f1 = (const float*)d_in[10];
    const float* Wih_r0 = (const float*)d_in[11], *Whh_r0 = (const float*)d_in[12];
    const float* bih_r0 = (const float*)d_in[13], *bhh_r0 = (const float*)d_in[14];
    const float* Wih_r1 = (const float*)d_in[15], *Whh_r1 = (const float*)d_in[16];
    const float* bih_r1 = (const float*)d_in[17], *bhh_r1 = (const float*)d_in[18];
    float* dout = (float*)d_out;

    // Workspace carve-out (~260 MB total)
    char* base = (char*)d_ws;
    size_t off = 0;
    auto alloc = [&](size_t bytes) {
        void* p = base + off;
        off += (bytes + 255) & ~(size_t)255;
        return p;
    };
    const size_t TB = (size_t)TSEQ * BATCH;
    bf16* emb_f = (bf16*)alloc(TB * NINP * 2);
    bf16* emb_r = (bf16*)alloc(TB * NINP * 2);
    bf16* xb_f  = (bf16*)alloc(TB * HID * 2);
    bf16* xb_r  = (bf16*)alloc(TB * HID * 2);
    bf16* Wc_f0 = (bf16*)alloc((size_t)GATES * (NINP + HID) * 2);
    bf16* Wc_r0 = (bf16*)alloc((size_t)GATES * (NINP + HID) * 2);
    bf16* Wc_f1 = (bf16*)alloc((size_t)GATES * (HID + HID) * 2);
    bf16* Wc_r1 = (bf16*)alloc((size_t)GATES * (HID + HID) * 2);
    float* b_f0 = (float*)alloc(GATES * 4);
    float* b_r0 = (float*)alloc(GATES * 4);
    float* b_f1 = (float*)alloc(GATES * 4);
    float* b_r1 = (float*)alloc(GATES * 4);
    bf16* hf0 = (bf16*)alloc((size_t)BATCH * HID * 2);
    bf16* hf1 = (bf16*)alloc((size_t)BATCH * HID * 2);
    bf16* hr0 = (bf16*)alloc((size_t)BATCH * HID * 2);
    bf16* hr1 = (bf16*)alloc((size_t)BATCH * HID * 2);
    float* cf = (float*)alloc((size_t)BATCH * HID * 4);
    float* cr = (float*)alloc((size_t)BATCH * HID * 4);

    // 1) embeddings (fwd + reversed) -> bf16
    embed_gather<<<TSEQ * BATCH, 128, 0, stream>>>(tok, lens, enc, emb_f, emb_r);

    // 2) pack fused [Wih | Whh] bf16 + fused biases
    pack_weights<<<1024, 256, 0, stream>>>(Wih_f0, Whh_f0, bih_f0, bhh_f0, Wc_f0, b_f0, NINP);
    pack_weights<<<1024, 256, 0, stream>>>(Wih_r0, Whh_r0, bih_r0, bhh_r0, Wc_r0, b_r0, NINP);
    pack_weights<<<1024, 256, 0, stream>>>(Wih_f1, Whh_f1, bih_f1, bhh_f1, Wc_f1, b_f1, HID);
    pack_weights<<<1024, 256, 0, stream>>>(Wih_r1, Whh_r1, bih_r1, bhh_r1, Wc_r1, b_r1, HID);

    const int stBlocks = (BATCH * HID + 255) / 256;
    dim3 grid(GATES / 64, 2);   // 64 unit-tiles x 2 directions

    // 3) layer 0: emb -> xb  (both directions per launch)
    init_state<<<stBlocks, 256, 0, stream>>>(hf0, hf1, hr0, hr1, cf, cr);
    for (int t = 0; t < TSEQ; ++t) {
        bf16 *hpf = (t & 1) ? hf1 : hf0, *hnf = (t & 1) ? hf0 : hf1;
        bf16 *hpr = (t & 1) ? hr1 : hr0, *hnr = (t & 1) ? hr0 : hr1;
        lstm_step<<<grid, 128, 0, stream>>>(emb_f, emb_r, Wc_f0, Wc_r0, b_f0, b_r0,
                                            hpf, hpr, hnf, hnr, cf, cr, lens,
                                            t, NINP, xb_f, xb_r, nullptr);
    }

    // 4) layer 1: xb -> d_out (fp32, concat f|r on last dim)
    init_state<<<stBlocks, 256, 0, stream>>>(hf0, hf1, hr0, hr1, cf, cr);
    for (int t = 0; t < TSEQ; ++t) {
        bf16 *hpf = (t & 1) ? hf1 : hf0, *hnf = (t & 1) ? hf0 : hf1;
        bf16 *hpr = (t & 1) ? hr1 : hr0, *hnr = (t & 1) ? hr0 : hr1;
        lstm_step<<<grid, 128, 0, stream>>>(xb_f, xb_r, Wc_f1, Wc_r1, b_f1, b_r1,
                                            hpf, hpr, hnf, hnr, cf, cr, lens,
                                            t, HID, nullptr, nullptr, dout);
    }
}
